// ConditionCrossAttention2D_24653112279454
// MI455X (gfx1250) — matrix operand order repaired and checked by reading, not compiled, not run on hardware
//
#include <hip/hip_runtime.h>
#include <hip/hip_bf16.h>

// ---------------------------------------------------------------------------
// ConditionCrossAttention2D for MI455X (gfx1250, wave32)
// WMMA f32_16x16x32_f16 flash-attention + Tensor Data Mover V staging.
// ---------------------------------------------------------------------------

typedef _Float16 h8   __attribute__((ext_vector_type(8)));
typedef _Float16 h16  __attribute__((ext_vector_type(16)));
typedef float    f8   __attribute__((ext_vector_type(8)));
typedef float    f4   __attribute__((ext_vector_type(4)));
typedef unsigned int u32x4 __attribute__((ext_vector_type(4)));
typedef int      i32x4 __attribute__((ext_vector_type(4)));
typedef int      i32x8 __attribute__((ext_vector_type(8)));

#define WMMA_F16(A, B, C) \
  __builtin_amdgcn_wmma_f32_16x16x32_f16(false, (A), false, (B), (short)0, (C), false, false)

static constexpr int BB = 4;     // batch
static constexpr int CF = 256;   // feature channels
static constexpr int CK = 32;    // key channels
static constexpr int NN = 4096;  // H*W
static constexpr int JB = 64;    // j-block rows staged per TDM transfer

// --------------------------- f32 -> f16 cast -------------------------------
__global__ void cca_cast_f16(const float* __restrict__ in,
                             _Float16* __restrict__ out, int n) {
  int i = blockIdx.x * blockDim.x + threadIdx.x;
  if (i < n) out[i] = (_Float16)in[i];
}

// --------------------------- projection GEMM -------------------------------
// D[m][n] = sum_c W[m][c] * X[b][c][n] + bias[m]
// colMajorOut==0: out[b][n][outC]  (q, v)   colMajorOut==1: out[b][m][N] (k)
__global__ void cca_proj_gemm(const _Float16* __restrict__ Wh,
                              const float* __restrict__ bias,
                              const _Float16* __restrict__ Xh,
                              _Float16* __restrict__ outp,
                              int outC, int colMajorOut) {
  const int lane = threadIdx.x & 31;
  const int wid  = threadIdx.x >> 5;
  const int job  = blockIdx.x * (blockDim.x >> 5) + wid;   // 0 .. B*256-1
  const int b    = job >> 8;
  const int n0   = (job & 255) * 16;
  const int Llo  = lane & 15;
  const int Lhi  = lane >> 4;
  const int baseK = 8 * Lhi;

  h16 bf[8];
#pragma unroll
  for (int ks = 0; ks < 8; ++ks) {
    for (int s = 0; s < 16; ++s) {
      const int kk = (s < 8) ? (baseK + s) : (8 + baseK + s);
      bf[ks][s] = Xh[((size_t)(b * CF + ks * 32 + kk)) * NN + n0 + Llo];
    }
  }

  const int nTiles = outC >> 4;
  for (int t = 0; t < nTiles; ++t) {
    f8 acc = {};
#pragma unroll
    for (int ks = 0; ks < 8; ++ks) {
      const _Float16* wr = Wh + (size_t)(t * 16 + Llo) * 256 + ks * 32;
      h8 lo = *(const h8*)(wr + baseK);
      h8 hi = *(const h8*)(wr + 16 + baseK);
      h16 a = __builtin_shufflevector(lo, hi, 0, 1, 2, 3, 4, 5, 6, 7,
                                      8, 9, 10, 11, 12, 13, 14, 15);
      acc = WMMA_F16(a, bf[ks], acc);
    }
#pragma unroll
    for (int r = 0; r < 8; ++r) acc[r] += bias[t * 16 + r + 8 * Lhi];

    if (colMajorOut) {
#pragma unroll
      for (int r = 0; r < 8; ++r) {
        int m = t * 16 + r + 8 * Lhi;
        outp[((size_t)(b * outC + m)) * NN + n0 + Llo] = (_Float16)acc[r];
      }
    } else {
      h8 ph;
#pragma unroll
      for (int r = 0; r < 8; ++r) ph[r] = (_Float16)acc[r];
      _Float16* dst =
          outp + ((size_t)(b * NN + n0 + Llo)) * outC + t * 16 + 8 * Lhi;
      *(h8*)dst = ph;
    }
  }
}

// ------------------------- TDM descriptor helper ---------------------------
// 2D tile: tile_dim0 = 256 f16 (contiguous channels), tile_dim1 = JB rows,
// tensor_dim0_stride = 256. data_size = 2 bytes. D# layout per ISA ch.8.
__device__ __forceinline__ void cca_tdm_load_v(unsigned int lds_addr,
                                               const _Float16* gptr) {
  unsigned long long ga = (unsigned long long)(size_t)gptr;
  u32x4 g0;
  g0[0] = 1u;                                   // count=1, user descriptor
  g0[1] = lds_addr;                             // lds_addr[31:0]
  g0[2] = (unsigned int)(ga & 0xFFFFFFFFu);     // global_addr[31:0]
  g0[3] = (unsigned int)((ga >> 32) & 0x01FFFFFFu) | 0x80000000u;  // type=2
  i32x8 g1;
  g1[0] = 0x00010000;          // workgroup_mask=0, data_size=1 (2 bytes)
  g1[1] = (int)(256u << 16);   // tensor_dim0 = 256 (bits [79:48], low half)
  g1[2] = (int)(4096u << 16);  // tensor_dim1 = 4096 (bits [111:80], low half)
  g1[3] = (int)(256u << 16);   // tile_dim0 = 256 (bits [127:112])
  g1[4] = JB;                  // tile_dim1 = 64, tile_dim2 = 0
  g1[5] = 256;                 // tensor_dim0_stride = 256 (bits [207:160])
  g1[6] = 0;
  g1[7] = 0;
  i32x4 g2 = {};               // 2D tensor: groups 2/3 unused
  i32x4 g3 = {};
  i32x8 g4 = {};               // extra group (6-arg toolchain form)
  __builtin_amdgcn_tensor_load_to_lds(g0, g1, g2, g3, g4, 0);
}

// ------------------------- flash attention kernel --------------------------
// 8 waves / workgroup; each wave owns a 16-row i-tile. V tiles (64 x 256 f16)
// are DMA'd to LDS by the Tensor Data Mover, double buffered, shared by all
// 8 waves. Online softmax over j-blocks of 64.
__global__ __launch_bounds__(256, 1)
void cca_attn(const _Float16* __restrict__ qh,   // [B][N][32]
              const _Float16* __restrict__ kh,   // [B][32][N]
              const _Float16* __restrict__ vh,   // [B][N][256]
              const float* __restrict__ feat,    // [B][256][N]
              const float* __restrict__ gammap,  // [1]
              float* __restrict__ outp)          // [B][256][N]
{
  __shared__ __align__(128) _Float16 Vtile[2][JB][CF];  // 2 x 32 KB
  __shared__ __align__(32)  _Float16 Pbuf[8][16][72];   // per-wave P re-layout

  const int lane = threadIdx.x & 31;
  const int wid  = threadIdx.x >> 5;
  const int job  = blockIdx.x * 8 + wid;  // 0 .. B*256-1
  const int b    = job >> 8;
  const int i0   = (job & 255) * 16;
  const int Llo  = lane & 15;
  const int Lhi  = lane >> 4;
  const int baseK = 8 * Lhi;
  const float g = gammap[0];

  // q A-fragment
  const _Float16* qrow = qh + ((size_t)(b * NN + i0 + Llo)) * CK;
  h8 qlo = *(const h8*)(qrow + baseK);
  h8 qhi = *(const h8*)(qrow + 16 + baseK);
  h16 aq = __builtin_shufflevector(qlo, qhi, 0, 1, 2, 3, 4, 5, 6, 7,
                                   8, 9, 10, 11, 12, 13, 14, 15);

  f8 acc[16];
#pragma unroll
  for (int t = 0; t < 16; ++t) acc[t] = (f8){};
  float m[8], l[8];
#pragma unroll
  for (int r = 0; r < 8; ++r) { m[r] = -3.0e38f; l[r] = 0.0f; }

  const _Float16* kbase = kh + ((size_t)(b * CK)) * NN + Llo;  // column j = Llo
  const _Float16* vbase = vh + ((size_t)b * NN) * CF;
  const f8 zero = {};
  const unsigned int vtile_lds[2] = {
      (unsigned int)(size_t)&Vtile[0][0][0],
      (unsigned int)(size_t)&Vtile[1][0][0]};

  // prologue: DMA first V tile into buffer 0 (one wave issues; EXEC ignored)
  if (wid == 0) cca_tdm_load_v(vtile_lds[0], vbase);

  const int nBlocks = NN / JB;  // 64
  for (int jb = 0; jb < nBlocks; ++jb) {
    const int j0  = jb * JB;
    const int cur = jb & 1;

    if (wid == 0) __builtin_amdgcn_s_wait_tensorcnt(0);
    __syncthreads();  // V tile `cur` visible to all waves
    if (wid == 0 && jb + 1 < nBlocks)
      cca_tdm_load_v(vtile_lds[cur ^ 1], vbase + (size_t)(j0 + JB) * CF);

    // ---- energy: 4 tiles of E[i][j], K = d_k = 32 ----
    if (jb + 1 < nBlocks) __builtin_prefetch(kbase + j0 + JB, 0, 3);
    h16 bk[4];
    f8  e[4];
#pragma unroll
    for (int t = 0; t < 4; ++t) {
      for (int s = 0; s < 16; ++s) {
        const int kk = (s < 8) ? (baseK + s) : (8 + baseK + s);
        bk[t][s] = kbase[(size_t)kk * NN + j0 + 16 * t];
      }
    }
#pragma unroll
    for (int t = 0; t < 4; ++t) e[t] = WMMA_F16(aq, bk[t], zero);

    // ---- online softmax over the 64-wide j-block ----
#pragma unroll
    for (int r = 0; r < 8; ++r) {
      float mx = fmaxf(fmaxf(e[0][r], e[1][r]), fmaxf(e[2][r], e[3][r]));
      mx = fmaxf(mx, __shfl_xor(mx, 1, 32));
      mx = fmaxf(mx, __shfl_xor(mx, 2, 32));
      mx = fmaxf(mx, __shfl_xor(mx, 4, 32));
      mx = fmaxf(mx, __shfl_xor(mx, 8, 32));
      float mn = fmaxf(m[r], mx);
      float p0 = __expf(e[0][r] - mn);
      float p1 = __expf(e[1][r] - mn);
      float p2 = __expf(e[2][r] - mn);
      float p3 = __expf(e[3][r] - mn);
      float rs = (p0 + p1) + (p2 + p3);
      rs += __shfl_xor(rs, 1, 32);
      rs += __shfl_xor(rs, 2, 32);
      rs += __shfl_xor(rs, 4, 32);
      rs += __shfl_xor(rs, 8, 32);
      float sc = __expf(m[r] - mn);
      l[r] = l[r] * sc + rs;
      m[r] = mn;
#pragma unroll
      for (int t = 0; t < 16; ++t) acc[t][r] *= sc;
      const int pr = r + 8 * Lhi;
      Pbuf[wid][pr][Llo]      = (_Float16)p0;
      Pbuf[wid][pr][Llo + 16] = (_Float16)p1;
      Pbuf[wid][pr][Llo + 32] = (_Float16)p2;
      Pbuf[wid][pr][Llo + 48] = (_Float16)p3;
    }
    asm volatile("s_wait_dscnt 0" ::: "memory");

    // P A-fragments (16 x 64 split into two 16 x 32)
    h8 pa = *(const h8*)&Pbuf[wid][Llo][baseK];
    h8 pb = *(const h8*)&Pbuf[wid][Llo][16 + baseK];
    h8 pc = *(const h8*)&Pbuf[wid][Llo][32 + baseK];
    h8 pd = *(const h8*)&Pbuf[wid][Llo][48 + baseK];
    h16 aP0 = __builtin_shufflevector(pa, pb, 0, 1, 2, 3, 4, 5, 6, 7,
                                      8, 9, 10, 11, 12, 13, 14, 15);
    h16 aP1 = __builtin_shufflevector(pc, pd, 0, 1, 2, 3, 4, 5, 6, 7,
                                      8, 9, 10, 11, 12, 13, 14, 15);

    // ---- PV from LDS V tile: column c = Llo, K-row j in table order ----
    const _Float16* vrow0 = &Vtile[cur][0][Llo];        // j = j0 + k
    const _Float16* vrow1 = &Vtile[cur][32][Llo];       // j = j0 + 32 + k
#pragma unroll
    for (int t = 0; t < 16; ++t) {
      h16 bv0, bv1;
      for (int s = 0; s < 16; ++s) {
        const int kk = (s < 8) ? (baseK + s) : (8 + baseK + s);
        bv0[s] = vrow0[kk * CF + t * 16];
        bv1[s] = vrow1[kk * CF + t * 16];
      }
      acc[t] = WMMA_F16(aP0, bv0, acc[t]);
      acc[t] = WMMA_F16(aP1, bv1, acc[t]);
    }
    __syncthreads();  // all waves done with tile `cur` before it is refilled
  }

  // ---- epilogue: normalize, out = gamma * attnout + features ----
  float inv[8];
#pragma unroll
  for (int r = 0; r < 8; ++r) inv[r] = 1.0f / l[r];

#pragma unroll
  for (int t = 0; t < 16; ++t) {
    int c = t * 16 + Llo;
    size_t base = ((size_t)(b * CF + c)) * NN + i0 + 8 * Lhi;
    f4 fa = *(const f4*)(feat + base);
    f4 fb = *(const f4*)(feat + base + 4);
    f4 oa, ob;
#pragma unroll
    for (int r = 0; r < 4; ++r) {
      oa[r] = g * (acc[t][r]     * inv[r])     + fa[r];
      ob[r] = g * (acc[t][r + 4] * inv[r + 4]) + fb[r];
    }
    *(f4*)(outp + base)     = oa;
    *(f4*)(outp + base + 4) = ob;
  }
}

// ------------------------------- launcher ----------------------------------
extern "C" void kernel_launch(void* const* d_in, const int* in_sizes, int n_in,
                              void* d_out, int out_size, void* d_ws,
                              size_t ws_size, hipStream_t stream) {
  (void)in_sizes; (void)n_in; (void)out_size; (void)ws_size;
  const float* features   = (const float*)d_in[0];
  const float* conditions = (const float*)d_in[1];
  const float* Wq = (const float*)d_in[2];
  const float* bq = (const float*)d_in[3];
  const float* Wk = (const float*)d_in[4];
  const float* bk = (const float*)d_in[5];
  const float* Wv = (const float*)d_in[6];
  const float* bv = (const float*)d_in[7];
  const float* gamma = (const float*)d_in[8];
  float* outp = (float*)d_out;

  char* ws = (char*)d_ws;
  size_t off = 0;
  auto alloc = [&](size_t bytes) -> void* {
    void* p = ws + off;
    off += (bytes + 255) & ~(size_t)255;
    return p;
  };
  _Float16* condh = (_Float16*)alloc((size_t)BB * CF * NN * 2);  // 8 MB
  _Float16* feath = (_Float16*)alloc((size_t)BB * CF * NN * 2);  // 8 MB
  _Float16* Wqh   = (_Float16*)alloc((size_t)CK * CF * 2);
  _Float16* Wkh   = (_Float16*)alloc((size_t)CK * CF * 2);
  _Float16* Wvh   = (_Float16*)alloc((size_t)CF * CF * 2);
  _Float16* qh    = (_Float16*)alloc((size_t)BB * NN * CK * 2);  // 1 MB
  _Float16* kh    = (_Float16*)alloc((size_t)BB * NN * CK * 2);  // 1 MB
  _Float16* vh    = (_Float16*)alloc((size_t)BB * NN * CF * 2);  // 8 MB

  auto cast = [&](const float* src, _Float16* dst, int n) {
    cca_cast_f16<<<(n + 255) / 256, 256, 0, stream>>>(src, dst, n);
  };
  cast(conditions, condh, BB * CF * NN);
  cast(features,   feath, BB * CF * NN);
  cast(Wq, Wqh, CK * CF);
  cast(Wk, Wkh, CK * CF);
  cast(Wv, Wvh, CF * CF);

  cca_proj_gemm<<<256, 128, 0, stream>>>(Wqh, bq, condh, qh, CK, 0);  // q [n][32]
  cca_proj_gemm<<<256, 128, 0, stream>>>(Wkh, bk, feath, kh, CK, 1);  // k [32][N]
  cca_proj_gemm<<<256, 128, 0, stream>>>(Wvh, bv, feath, vh, CF, 0);  // v [n][256]

  cca_attn<<<128, 256, 0, stream>>>(qh, kh, vh, features, gamma, outp);
}
